// BilinearRelationNet_82308753261052
// MI455X (gfx1250) — matrix operand order repaired and checked by reading, not compiled, not run on hardware
//
#include <hip/hip_runtime.h>
#include <hip/hip_bf16.h>

typedef __attribute__((ext_vector_type(16))) __bf16 v16bf;
typedef __attribute__((ext_vector_type(8)))  float  v8f;

#define N_ROWS 65536
#define D_IN   512
#define H1     256
#define H2     128
#define H3     64
#define TILE_M 128
#define KC     64
#define LDX    72    // padded LDS stride (bf16 elems) for 64-wide K buffers: 144B, 16B multiple
#define LDE    136   // padded LDS stride for 128-wide K buffers: 272B, 16B multiple

static __device__ __forceinline__ v8f wmma_bf16(v16bf a, v16bf b, v8f c) {
  // D = A(16x32 bf16) * B(32x16 bf16) + C(16x16 f32)
  return __builtin_amdgcn_wmma_f32_16x16x32_bf16(false, a, false, b, (short)0, c, false, false);
}

// A-matrix fragment (16x32 bf16), LDS row-major [row][k], stride in elems.
// Lane r=lane&15 -> row r. half=lane>>4 -> K base 8*half; elems 8..15 -> K+16.
static __device__ __forceinline__ v16bf ldfragA(const __bf16* base, int stride, int lane) {
  const __bf16* p = base + (lane & 15) * stride + ((lane >> 4) << 3);
  v16bf f;
  ((uint4*)&f)[0] = *(const uint4*)(p);
  ((uint4*)&f)[1] = *(const uint4*)(p + 16);
  return f;
}
// B-matrix fragment (32x16 bf16) from LDS holding B^T row-major [n][k].
// Lane n=lane&15 -> col n. half=lane>>4 -> K base 8*half; elems 8..15 -> K+16.
static __device__ __forceinline__ v16bf ldfragB(const __bf16* base, int stride, int lane) {
  const __bf16* p = base + (lane & 15) * stride + ((lane >> 4) << 3);
  v16bf f;
  ((uint4*)&f)[0] = *(const uint4*)(p);
  ((uint4*)&f)[1] = *(const uint4*)(p + 16);
  return f;
}

__global__ void k_zero_stats(float* __restrict__ stats) {
  int t = blockIdx.x * blockDim.x + threadIdx.x;
  if (t < 1024) stats[t] = 0.f;
}

// GEMM1: h_pre = x @ W1 + b1 (bf16 WMMA, f32 accum), store h_pre as bf16,
// accumulate per-column sum / sum-of-squares for BatchNorm statistics.
// Block: 256 threads = 8 waves. Tile: 128 rows x 256 cols, K staged in 64-chunks.
// Wave (wm = w&3, wn = w>>2): rows 32*wm..+32, cols 128*wn..+128.
__global__ __launch_bounds__(256) void k_gemm1_stats(
    const float* __restrict__ x1, const float* __restrict__ x2,
    const float* __restrict__ W1, const float* __restrict__ b1,
    __bf16* __restrict__ hpre, float* __restrict__ stats)
{
  __shared__ __align__(16) __bf16 xa[TILE_M * LDX];
  __shared__ __align__(16) __bf16 wt[H1 * LDX];
  __shared__ float csum[H1];
  __shared__ float csq[H1];

  const int t    = threadIdx.x;
  const int lane = t & 31;
  const int w    = t >> 5;
  const int wm   = w & 3;
  const int wn   = w >> 2;
  const int r    = lane & 15;
  const int half = lane >> 4;
  const int which = blockIdx.y;
  const float* __restrict__ x = which ? x2 : x1;
  __bf16* __restrict__ hp = hpre + (size_t)which * (size_t)N_ROWS * H1;
  float* __restrict__ st  = stats + which * 2 * H1;
  const size_t row0 = (size_t)blockIdx.x * TILE_M;

  v8f acc[2][8];
  const v8f z = {0.f,0.f,0.f,0.f,0.f,0.f,0.f,0.f};
  for (int s = 0; s < 2; ++s)
    for (int n = 0; n < 8; ++n) acc[s][n] = z;
  csum[t] = 0.f; csq[t] = 0.f;

  for (int kc = 0; kc < D_IN / KC; ++kc) {
    __syncthreads();
    // Stage x tile: 128 rows x 64 K, f32 -> bf16 LDS (row-major, stride LDX)
    for (int i = 0; i < 8; ++i) {
      int l = i * 256 + t;
      int row = l >> 4, c4 = l & 15;
      float4 v = *(const float4*)(x + (row0 + row) * D_IN + kc * KC + c4 * 4);
      __bf16* d = xa + row * LDX + c4 * 4;
      d[0] = (__bf16)v.x; d[1] = (__bf16)v.y; d[2] = (__bf16)v.z; d[3] = (__bf16)v.w;
    }
    // Stage W1 chunk (64 K x 256 N) transposed -> wt[n][k]
    for (int i = 0; i < 16; ++i) {
      int l = i * 256 + t;
      int k = l >> 6, n4 = l & 63;
      float4 v = *(const float4*)(W1 + (size_t)(kc * KC + k) * H1 + n4 * 4);
      wt[(n4 * 4 + 0) * LDX + k] = (__bf16)v.x;
      wt[(n4 * 4 + 1) * LDX + k] = (__bf16)v.y;
      wt[(n4 * 4 + 2) * LDX + k] = (__bf16)v.z;
      wt[(n4 * 4 + 3) * LDX + k] = (__bf16)v.w;
    }
    if (kc + 1 < D_IN / KC) {  // hint next x chunk into cache (global_prefetch)
      __builtin_prefetch(x + (row0 + (t >> 1)) * D_IN + (kc + 1) * KC + (t & 1) * 32, 0, 1);
    }
    __syncthreads();
    for (int kk = 0; kk < 2; ++kk) {
      v16bf a0 = ldfragA(xa + (32 * wm     ) * LDX + kk * 32, LDX, lane);
      v16bf a1 = ldfragA(xa + (32 * wm + 16) * LDX + kk * 32, LDX, lane);
      for (int n = 0; n < 8; ++n) {
        v16bf b = ldfragB(wt + (128 * wn + 16 * n) * LDX + kk * 32, LDX, lane);
        acc[0][n] = wmma_bf16(a0, b, acc[0][n]);
        acc[1][n] = wmma_bf16(a1, b, acc[1][n]);
      }
    }
  }

  // Epilogue: +b1, store bf16 h_pre, accumulate column stats.
  // C layout: lane r -> col; VGPR i -> row i + 8*half.
  for (int s = 0; s < 2; ++s) {
    for (int n = 0; n < 8; ++n) {
      int col = 128 * wn + 16 * n + r;
      float bb = b1[col];
      float ls = 0.f, lq = 0.f;
      size_t rb = row0 + 32 * wm + 16 * s + 8 * half;
      for (int i = 0; i < 8; ++i) {
        float v = acc[s][n][i] + bb;
        hp[(rb + i) * H1 + col] = (__bf16)v;
        ls += v; lq += v * v;
      }
      atomicAdd(&csum[col], ls);
      atomicAdd(&csq[col], lq);
    }
  }
  __syncthreads();
  atomicAdd(&st[t], csum[t]);
  atomicAdd(&st[H1 + t], csq[t]);
}

// Fused tail: BN+ReLU -> GEMM2 (both towers) -> cosine sim -> combined
// features -> GEMM3 -> ReLU -> dot(W4) -> sigmoid -> blend/clip.
// Block: 256 threads = 8 waves; tile 128 rows. Wave w owns rows 16w..16w+15.
__global__ __launch_bounds__(256) void k_fused_tail(
    const __bf16* __restrict__ hpre, const float* __restrict__ stats,
    const float* __restrict__ gamma, const float* __restrict__ beta_bn,
    const float* __restrict__ W2, const float* __restrict__ b2,
    const float* __restrict__ W3, const float* __restrict__ b3,
    const float* __restrict__ W4, const float* __restrict__ b4,
    const float* __restrict__ alpha, const float* __restrict__ beta,
    float* __restrict__ out)
{
  __shared__ __align__(16) unsigned char smem[55296];
  __shared__ float bn_a[2][H1];
  __shared__ float bn_c[2][H1];

  __bf16* h1c  = (__bf16*)smem;              // phase A: 128 x LDX
  __bf16* h2c  = (__bf16*)(smem + 18432);    // phase A: 128 x LDX
  __bf16* wbuf = (__bf16*)(smem + 36864);    // W2^T [128][LDX] or W3^T [64][LDE]
  __bf16* ebuf = (__bf16*)smem;              // phase B: e-part [128][LDE] (unions h1c/h2c)

  const int t    = threadIdx.x;
  const int lane = t & 31;
  const int w    = t >> 5;
  const int r    = lane & 15;
  const int half = lane >> 4;
  const size_t row0 = (size_t)blockIdx.x * TILE_M;

  // Finalize BN params: y = a*h + c, a = gamma*rsqrt(var+eps), c = beta - a*mean
  for (int which = 0; which < 2; ++which) {
    float s = stats[which * 2 * H1 + t];
    float q = stats[which * 2 * H1 + H1 + t];
    float mean = s * (1.f / N_ROWS);
    float var  = q * (1.f / N_ROWS) - mean * mean;
    float a = gamma[t] * rsqrtf(var + 1e-5f);
    bn_a[which][t] = a;
    bn_c[which][t] = beta_bn[t] - a * mean;
  }

  const __bf16* hp1 = hpre;
  const __bf16* hp2 = hpre + (size_t)N_ROWS * H1;

  v8f acc1[8], acc2[8];
  const v8f z = {0.f,0.f,0.f,0.f,0.f,0.f,0.f,0.f};
  for (int n = 0; n < 8; ++n) { acc1[n] = z; acc2[n] = z; }

  // ---- Phase A: h = relu(BN(h_pre)); towers @ W2 ----
  for (int kc = 0; kc < H1 / KC; ++kc) {
    __syncthreads();
    // Stage BN+ReLU'd activation chunks (128 rows x 64 K) for both towers
    for (int i = 0; i < 4; ++i) {
      int l = i * 256 + t;
      int row = l >> 3, c8 = l & 7;
      int colb = kc * KC + c8 * 8;
      uint4 u1 = *(const uint4*)(hp1 + (size_t)(row0 + row) * H1 + colb);
      uint4 u2 = *(const uint4*)(hp2 + (size_t)(row0 + row) * H1 + colb);
      __bf16 e1[8], e2[8], o1[8], o2[8];
      *(uint4*)e1 = u1; *(uint4*)e2 = u2;
      for (int j = 0; j < 8; ++j) {
        int col = colb + j;
        o1[j] = (__bf16)fmaxf(bn_a[0][col] * (float)e1[j] + bn_c[0][col], 0.f);
        o2[j] = (__bf16)fmaxf(bn_a[1][col] * (float)e2[j] + bn_c[1][col], 0.f);
      }
      *(uint4*)(h1c + row * LDX + c8 * 8) = *(uint4*)o1;
      *(uint4*)(h2c + row * LDX + c8 * 8) = *(uint4*)o2;
    }
    // Stage W2 chunk (64 K x 128 N) transposed -> wbuf[n][k]
    for (int i = 0; i < 8; ++i) {
      int l = i * 256 + t;
      int k = l >> 5, n4 = l & 31;
      float4 v = *(const float4*)(W2 + (size_t)(kc * KC + k) * H2 + n4 * 4);
      wbuf[(n4 * 4 + 0) * LDX + k] = (__bf16)v.x;
      wbuf[(n4 * 4 + 1) * LDX + k] = (__bf16)v.y;
      wbuf[(n4 * 4 + 2) * LDX + k] = (__bf16)v.z;
      wbuf[(n4 * 4 + 3) * LDX + k] = (__bf16)v.w;
    }
    __syncthreads();
    for (int kk = 0; kk < 2; ++kk) {
      v16bf a1 = ldfragA(h1c + (16 * w) * LDX + kk * 32, LDX, lane);
      v16bf a2 = ldfragA(h2c + (16 * w) * LDX + kk * 32, LDX, lane);
      for (int n = 0; n < 8; ++n) {
        v16bf b = ldfragB(wbuf + (16 * n) * LDX + kk * 32, LDX, lane);
        acc1[n] = wmma_bf16(a1, b, acc1[n]);
        acc2[n] = wmma_bf16(a2, b, acc2[n]);
      }
    }
  }

  // +b2, ReLU, and row-wise dot/norm reductions in C-fragment layout.
  float d[8], s1[8], s2[8];
  for (int i = 0; i < 8; ++i) { d[i] = 0.f; s1[i] = 0.f; s2[i] = 0.f; }
  for (int n = 0; n < 8; ++n) {
    int col = 16 * n + r;
    float bb = b2[col];
    for (int i = 0; i < 8; ++i) {
      float a  = fmaxf(acc1[n][i] + bb, 0.f);
      float b_ = fmaxf(acc2[n][i] + bb, 0.f);
      acc1[n][i] = a; acc2[n][i] = b_;
      d[i] += a * b_; s1[i] += a * a; s2[i] += b_ * b_;
    }
  }
  for (int off = 1; off < 16; off <<= 1) {
    for (int i = 0; i < 8; ++i) {
      d[i]  += __shfl_xor(d[i],  off, 16);
      s1[i] += __shfl_xor(s1[i], off, 16);
      s2[i] += __shfl_xor(s2[i], off, 16);
    }
  }
  float sm[8];
  for (int i = 0; i < 8; ++i) {
    float inv1 = 1.f / fmaxf(sqrtf(s1[i]), 1e-15f);
    float inv2 = 1.f / fmaxf(sqrtf(s2[i]), 1e-15f);
    sm[i] = fminf(fmaxf(d[i] * inv1 * inv2, 0.f), 1.f);
  }

  // ---- Phase B: combined = [h1*h2, |h1-h2|, h1+h2] @ W3 as split-K ----
  v8f acc3[4];
  for (int n = 0; n < 4; ++n) acc3[n] = z;
  for (int p = 0; p < 3; ++p) {
    __syncthreads();
    // Each wave writes its 16 rows of the e-part (C layout -> row-major LDS)
    for (int n = 0; n < 8; ++n) {
      int col = 16 * n + r;
      for (int i = 0; i < 8; ++i) {
        int row = 16 * w + 8 * half + i;
        float a = acc1[n][i], b_ = acc2[n][i];
        float v = (p == 0) ? a * b_ : (p == 1) ? fabsf(a - b_) : (a + b_);
        ebuf[row * LDE + col] = (__bf16)v;
      }
    }
    // Stage W3 part (rows 128p..+128, 64 cols) transposed -> wbuf[n][k]
    for (int i = 0; i < 8; ++i) {
      int l = i * 256 + t;
      int k = l >> 4, n4 = l & 15;
      float4 v = *(const float4*)(W3 + (size_t)(128 * p + k) * H3 + n4 * 4);
      wbuf[(n4 * 4 + 0) * LDE + k] = (__bf16)v.x;
      wbuf[(n4 * 4 + 1) * LDE + k] = (__bf16)v.y;
      wbuf[(n4 * 4 + 2) * LDE + k] = (__bf16)v.z;
      wbuf[(n4 * 4 + 3) * LDE + k] = (__bf16)v.w;
    }
    __syncthreads();
    for (int kk = 0; kk < 4; ++kk) {
      v16bf a = ldfragA(ebuf + (16 * w) * LDE + kk * 32, LDE, lane);
      for (int n = 0; n < 4; ++n) {
        v16bf b = ldfragB(wbuf + (16 * n) * LDE + kk * 32, LDE, lane);
        acc3[n] = wmma_bf16(a, b, acc3[n]);
      }
    }
  }

  // s_learn = relu(acc3 + b3) . W4 + b4 ; final = clip(alpha*sm + beta*sigmoid)
  float sl[8];
  for (int i = 0; i < 8; ++i) sl[i] = 0.f;
  for (int n = 0; n < 4; ++n) {
    int col = 16 * n + r;
    float b3v = b3[col];
    float w4v = W4[col];
    for (int i = 0; i < 8; ++i)
      sl[i] += fmaxf(acc3[n][i] + b3v, 0.f) * w4v;
  }
  for (int off = 1; off < 16; off <<= 1)
    for (int i = 0; i < 8; ++i) sl[i] += __shfl_xor(sl[i], off, 16);

  float al = alpha[0], be = beta[0], b4v = b4[0];
  if (r == 0) {
    for (int i = 0; i < 8; ++i) {
      float xx  = sl[i] + b4v;
      float sig = 1.f / (1.f + expf(-xx));
      float f   = al * sm[i] + be * sig;
      out[row0 + 16 * w + 8 * half + i] = fminf(fmaxf(f, 0.f), 1.f);
    }
  }
}

extern "C" void kernel_launch(void* const* d_in, const int* in_sizes, int n_in,
                              void* d_out, int out_size, void* d_ws, size_t ws_size,
                              hipStream_t stream) {
  (void)in_sizes; (void)n_in; (void)out_size; (void)ws_size;
  const float* x1      = (const float*)d_in[0];
  const float* x2      = (const float*)d_in[1];
  const float* W1      = (const float*)d_in[2];
  const float* b1      = (const float*)d_in[3];
  const float* gamma   = (const float*)d_in[4];
  const float* beta_bn = (const float*)d_in[5];
  const float* W2      = (const float*)d_in[6];
  const float* b2      = (const float*)d_in[7];
  const float* W3      = (const float*)d_in[8];
  const float* b3      = (const float*)d_in[9];
  const float* W4      = (const float*)d_in[10];
  const float* b4      = (const float*)d_in[11];
  const float* alpha   = (const float*)d_in[12];
  const float* beta    = (const float*)d_in[13];
  float* out = (float*)d_out;

  // Workspace: h_pre (both towers, bf16) + 1024 floats of BN stats (~67 MB)
  __bf16* hpre = (__bf16*)d_ws;
  float* stats = (float*)(hpre + (size_t)2 * N_ROWS * H1);

  k_zero_stats<<<dim3(4), dim3(256), 0, stream>>>(stats);
  k_gemm1_stats<<<dim3(N_ROWS / TILE_M, 2), dim3(256), 0, stream>>>(
      x1, x2, W1, b1, hpre, stats);
  k_fused_tail<<<dim3(N_ROWS / TILE_M), dim3(256), 0, stream>>>(
      hpre, stats, gamma, beta_bn, W2, b2, W3, b3, W4, b4, alpha, beta, out);
}
